// FFF_86122684220310
// MI455X (gfx1250) — matrix operand order repaired and checked by reading, not compiled, not run on hardware
//
#include <hip/hip_runtime.h>

// Fast-FeedForward tree network for MI455X (gfx1250).
//   x: [32768,1024] f32   w1s: [2047,1024] f32   w2s: [2047,1024] f32   y: [32768,1024] f32
//
// Bound by memory: ~256MB HBM + ~2GB intrinsic L2 gather traffic (each row
// touches 11 w1 rows + 11 w2 rows; weights + bf16-staged x are L2-resident).
// One wave32 owns a 16-row tile; scores via v_wmma_f32_16x16x32_bf16
// (A = x tile rows, B = 16 gathered w1 columns, keep diagonal of D, f32 accum).
// Fast path pre-converts x and w1s to bf16 in d_ws (halves score-phase L2
// traffic; inner loop = 4x b128 loads + 1 wmma). Fallback converts on the fly
// if ws_size is too small. Output pass streams gathered w2s rows (f32) with
// coalesced loads and non-temporal stores (y is write-once; keep it out of L2
// so the gather working set stays resident).

#define N_IN    1024
#define N_OUT   1024
#define DEPTH   10
#define LEVELS  (DEPTH + 1)
#define BATCH   32768
#define N_NODES 2047

typedef __attribute__((ext_vector_type(16))) __bf16 v16bf;
typedef __attribute__((ext_vector_type(8)))  __bf16 v8bf;
typedef __attribute__((ext_vector_type(8)))  float  v8f;
typedef __attribute__((ext_vector_type(4)))  float  v4f;   // native vector: OK for NT builtins

__device__ __forceinline__ v4f ldg4(const float* p) {
    return *reinterpret_cast<const v4f*>(p);
}

// Branchless diagonal extraction from the 16x16 f32 C fragment.
// C layout: VGPR j holds M = j + hi*8, N = m. Each lane selects acc[m&7]
// (cndmask chain) giving D[hi*8 + (m&7)][m]; the true diagonal of row m sits
// in lane (m<8 ? m : m+16) -> one ds_bpermute.
__device__ __forceinline__ float diag_extract(const v8f& acc, int jsel, int srcLane) {
    float val = acc[0];
    #pragma unroll
    for (int j = 1; j < 8; ++j)
        val = (jsel == j) ? acc[j] : val;
    return __shfl(val, srcLane, 32);
}

// Phase 2: y[row] = sum_l s_l * w2s[node_l]. Nodes are recomputed from the
// broadcast score signs (node_{l+1} = 2*node_l + (s_l>0) + 1), so no node
// array (and no movrel waterfall for it) is needed in phase 1.
__device__ __forceinline__ void phase2(int tile, int lane,
                                       const float* __restrict__ scores,
                                       const float* __restrict__ w2s,
                                       float* __restrict__ y) {
    const int lane4 = lane * 4;
    for (int r = 0; r < 16; ++r) {
        float sB[LEVELS];
        int   nB[LEVELS];
        int   n = 0;
        #pragma unroll
        for (int l = 0; l < LEVELS; ++l) {
            sB[l] = __shfl(scores[l], r, 32);
            nB[l] = n;
            n = n * 2 + (sB[l] > 0.0f ? 1 : 0) + 1;
        }
        float* yrow = y + (size_t)(tile * 16 + r) * N_OUT;
        #pragma unroll 2
        for (int cb = 0; cb < N_OUT / 128; ++cb) {
            const int j = cb * 128 + lane4;
            v4f acc4 = {0.f, 0.f, 0.f, 0.f};
            #pragma unroll
            for (int l = 0; l < LEVELS; ++l) {
                const v4f w = ldg4(w2s + (size_t)nB[l] * N_OUT + j);
                acc4[0] += sB[l] * w[0];
                acc4[1] += sB[l] * w[1];
                acc4[2] += sB[l] * w[2];
                acc4[3] += sB[l] * w[3];
            }
            // y is write-once: non-temporal store, don't pollute L2.
            __builtin_nontemporal_store(acc4, reinterpret_cast<v4f*>(yrow + j));
        }
    }
}

// ---------------- Prep: f32 -> bf16 (RNE via v_cvt), 8 elems/thread ----------------
// Source is read exactly once and the bf16 copy is the only consumer ->
// non-temporal loads; regular stores so the bf16 copy lands in L2.
__global__ __launch_bounds__(256)
void FFF_cvt_bf16_kernel(const float* __restrict__ in, __bf16* __restrict__ out, int n8) {
    const int i = blockIdx.x * blockDim.x + threadIdx.x;
    if (i >= n8) return;
    const v4f* p = reinterpret_cast<const v4f*>(in) + (size_t)i * 2;
    const v4f f0 = __builtin_nontemporal_load(p);
    const v4f f1 = __builtin_nontemporal_load(p + 1);
    v8bf o;
    o[0] = (__bf16)f0[0]; o[1] = (__bf16)f0[1]; o[2] = (__bf16)f0[2]; o[3] = (__bf16)f0[3];
    o[4] = (__bf16)f1[0]; o[5] = (__bf16)f1[1]; o[6] = (__bf16)f1[2]; o[7] = (__bf16)f1[3];
    reinterpret_cast<v8bf*>(out)[i] = o;
}

// ---------------- Fast path: bf16-staged inputs ----------------
__global__ __launch_bounds__(256, 2)
void FFF_86122684220310_bf16(const __bf16* __restrict__ xb,
                             const __bf16* __restrict__ w1b,
                             const float*  __restrict__ w2s,
                             float*        __restrict__ y) {
    const int lane = threadIdx.x & 31;
    const int wave = threadIdx.x >> 5;
    const int m    = lane & 15;
    const int hi   = lane >> 4;
    const int tile = blockIdx.x * 8 + wave;
    const int row  = tile * 16 + m;

    const __bf16* xrow = xb + (size_t)row * N_IN;
    const int jsel    = m & 7;
    const int srcLane = m + ((m & 8) << 1);

    float scores[LEVELS];
    int   node = 0;

    for (int l = 0; l < LEVELS; ++l) {
        const __bf16* w1 = w1b + (size_t)node * N_IN;

        v8f acc = {0.f, 0.f, 0.f, 0.f, 0.f, 0.f, 0.f, 0.f};

        #pragma unroll 4
        for (int c = 0; c < N_IN / 32; ++c) {
            // A (16-bit A 16x32): elems 0..7 -> K = c*32+hi*8+{0..7}, 8..15 -> +16.
            const int ka = c * 32 + hi * 8;
            const v8bf a0 = *reinterpret_cast<const v8bf*>(xrow + ka);        // b128
            const v8bf a1 = *reinterpret_cast<const v8bf*>(xrow + ka + 16);   // b128
            // B (16-bit B 32x16): elems 0..7 -> K = c*32+hi*8+{0..7}, 8..15 -> +16.
            const int kb = c * 32 + hi * 8;
            const v8bf b0 = *reinterpret_cast<const v8bf*>(w1 + kb);          // b128
            const v8bf b1 = *reinterpret_cast<const v8bf*>(w1 + kb + 16);     // b128

            const v16bf A = __builtin_shufflevector(a0, a1, 0,1,2,3,4,5,6,7,8,9,10,11,12,13,14,15);
            const v16bf B = __builtin_shufflevector(b0, b1, 0,1,2,3,4,5,6,7,8,9,10,11,12,13,14,15);
            acc = __builtin_amdgcn_wmma_f32_16x16x32_bf16(
                false, A, false, B, (short)0, acc, false, false);
        }

        const float s = diag_extract(acc, jsel, srcLane);
        scores[l] = s;
        node = node * 2 + (s > 0.0f ? 1 : 0) + 1;      // matches reference recurrence
    }

    phase2(tile, lane, scores, w2s, y);
}

// ---------------- Fallback: convert f32 -> bf16 on the fly ----------------
__global__ __launch_bounds__(256, 2)
void FFF_86122684220310_f32(const float* __restrict__ x,
                            const float* __restrict__ w1s,
                            const float* __restrict__ w2s,
                            float*       __restrict__ y) {
    const int lane = threadIdx.x & 31;
    const int wave = threadIdx.x >> 5;
    const int m    = lane & 15;
    const int hi   = lane >> 4;
    const int tile = blockIdx.x * 8 + wave;
    const int row  = tile * 16 + m;

    const float* xrow = x + (size_t)row * N_IN;
    const int jsel    = m & 7;
    const int srcLane = m + ((m & 8) << 1);

    float scores[LEVELS];
    int   node = 0;

    for (int l = 0; l < LEVELS; ++l) {
        const float* w1 = w1s + (size_t)node * N_IN;

        v8f acc = {0.f, 0.f, 0.f, 0.f, 0.f, 0.f, 0.f, 0.f};

        #pragma unroll 2
        for (int c = 0; c < N_IN / 32; ++c) {
            const int ka = c * 32 + hi * 8;
            float fa[16];
            *reinterpret_cast<v4f*>(&fa[0])  = ldg4(xrow + ka);
            *reinterpret_cast<v4f*>(&fa[4])  = ldg4(xrow + ka + 4);
            *reinterpret_cast<v4f*>(&fa[8])  = ldg4(xrow + ka + 16);
            *reinterpret_cast<v4f*>(&fa[12]) = ldg4(xrow + ka + 20);
            const int kb = c * 32 + hi * 8;
            float fb[16];
            *reinterpret_cast<v4f*>(&fb[0])  = ldg4(w1 + kb);
            *reinterpret_cast<v4f*>(&fb[4])  = ldg4(w1 + kb + 4);
            *reinterpret_cast<v4f*>(&fb[8])  = ldg4(w1 + kb + 16);
            *reinterpret_cast<v4f*>(&fb[12]) = ldg4(w1 + kb + 20);

            v16bf A, B;
            #pragma unroll
            for (int e = 0; e < 16; ++e) {
                A[e] = (__bf16)fa[e];
                B[e] = (__bf16)fb[e];
            }
            acc = __builtin_amdgcn_wmma_f32_16x16x32_bf16(
                false, A, false, B, (short)0, acc, false, false);
        }

        const float s = diag_extract(acc, jsel, srcLane);
        scores[l] = s;
        node = node * 2 + (s > 0.0f ? 1 : 0) + 1;
    }

    phase2(tile, lane, scores, w2s, y);
}

extern "C" void kernel_launch(void* const* d_in, const int* in_sizes, int n_in,
                              void* d_out, int out_size, void* d_ws, size_t ws_size,
                              hipStream_t stream) {
    (void)in_sizes; (void)n_in; (void)out_size;
    const float* x   = (const float*)d_in[0];
    const float* w1s = (const float*)d_in[1];
    const float* w2s = (const float*)d_in[2];
    float*       y   = (float*)d_out;

    const int blocks = (BATCH / 16) / 8;   // 2048 tiles, 8 waves/block -> 256 blocks

    const size_t xb_bytes  = (size_t)BATCH * N_IN * sizeof(__bf16);    // 64 MB
    const size_t w1b_bytes = (size_t)N_NODES * N_IN * sizeof(__bf16);  // ~4 MB

    if (ws_size >= xb_bytes + w1b_bytes) {
        // Fast path: stage bf16 copies of x and w1s in workspace, then run the
        // pure-bf16 WMMA kernel (4x b128 loads + 1 wmma per K-chunk, no cvts).
        __bf16* xb  = (__bf16*)d_ws;
        __bf16* w1b = (__bf16*)((char*)d_ws + xb_bytes);

        const int x_n8  = BATCH * N_IN / 8;        // 4,194,304
        const int w1_n8 = N_NODES * N_IN / 8;      // 262,016
        FFF_cvt_bf16_kernel<<<dim3((x_n8 + 255) / 256), dim3(256), 0, stream>>>(x, xb, x_n8);
        FFF_cvt_bf16_kernel<<<dim3((w1_n8 + 255) / 256), dim3(256), 0, stream>>>(w1s, w1b, w1_n8);
        FFF_86122684220310_bf16<<<dim3(blocks), dim3(256), 0, stream>>>(xb, w1b, w2s, y);
    } else {
        // Fallback: identical math, f32 loads + on-the-fly v_cvt_pk_bf16_f32.
        FFF_86122684220310_f32<<<dim3(blocks), dim3(256), 0, stream>>>(x, w1s, w2s, y);
    }
}
